// KnrmKGRanker_4105988735735
// MI455X (gfx1250) — matrix operand order repaired and checked by reading, not compiled, not run on hardware
//
#include <hip/hip_runtime.h>

typedef _Float16 v16h __attribute__((ext_vector_type(16)));
typedef _Float16 v8h  __attribute__((ext_vector_type(8)));
typedef float    v8f  __attribute__((ext_vector_type(8)));
typedef int      v4i  __attribute__((ext_vector_type(4)));

#define WORD_DIM 300
#define ENT_DIM  128
#define ATT_DIM  256
#define KPAD     448          // 300 + 128 padded to multiple of 32
#define B_SZ     64
#define QLEN     32
#define DLEN     2048
#define NKERN    11
#define TM       64           // transform rows per workgroup (4x M-tiles per wave)

// Build a 16x32 f16 A-fragment for V_WMMA_F32_16X16X32_F16 from a row-major row.
// Lanes 0-15 hold M=lane, K = {0..15}; lanes 16-31: K = {16..31} (matches B fragment).
// Caller passes p = &row[k0 + kSel*16]; we read [p, p+16).
__device__ __forceinline__ v16h make_a_frag(const _Float16* p) {
  v8h lo = *(const v8h*)p;
  v8h hi = *(const v8h*)(p + 8);
  v16h a;
#pragma unroll
  for (int i = 0; i < 8; ++i) { a[i] = lo[i]; a[8 + i] = hi[i]; }
  return a;
}

// ---------------------------------------------------------------------------
// Kernel 1: combined f16 weight matrix, transposed (N-major) + fused bias.
// WcT[n][k] = k<300 ? W_t[k][n] : (k<428 ? W_e[k-300][n] : 0);  bias = b_t+b_e
// ---------------------------------------------------------------------------
__global__ void prep_weights(const float* __restrict__ Wt, const float* __restrict__ bt,
                             const float* __restrict__ We, const float* __restrict__ be,
                             _Float16* __restrict__ WcT, float* __restrict__ bias) {
  int i = blockIdx.x * blockDim.x + threadIdx.x;   // [0, 448*256)
  int k = i >> 8;
  int n = i & 255;
  float v = 0.f;
  if (k < WORD_DIM)                v = Wt[k * ATT_DIM + n];
  else if (k < WORD_DIM + ENT_DIM) v = We[(k - WORD_DIM) * ATT_DIM + n];
  WcT[(size_t)n * KPAD + k] = (_Float16)v;
  if (i < ATT_DIM) bias[i] = bt[i] + be[i];
}

// ---------------------------------------------------------------------------
// Kernel 2: fused embed-gather + [64 x 448] @ [448 x 256] + bias + relu -> f16
// Workgroup = 64 output rows (57KB LDS A tile); 8 waves, each wave owns a
// 32-wide N-slab and a 4-high stack of M-tiles => 4x2 WMMA tiles, 112 WMMA/wave.
// B (weights) is loaded once per k-step and reused across 4 M-tiles.
// ---------------------------------------------------------------------------
__global__ void __launch_bounds__(256) transform_kernel(
    const int* __restrict__ tok, const float* __restrict__ ent,
    const float* __restrict__ embed, const _Float16* __restrict__ WcT,
    const float* __restrict__ bias, _Float16* __restrict__ out) {
  __shared__ _Float16 As[TM * KPAD];          // 57344 B
  const int rowBase = blockIdx.x * TM;
  const int t = threadIdx.x;

  // ---- stage A (f32 -> f16), vectorized with float4 loads ----
  // word part: 64 rows x 75 float4 (embed rows are 16B aligned: 300*4B = 1200B)
  for (int idx = t; idx < TM * 75; idx += 256) {
    int r = idx / 75, j = idx - r * 75;
    long tk = (long)tok[rowBase + r] + 1;      // -1 padding -> row 0
    const float4 w = *(const float4*)&embed[tk * WORD_DIM + j * 4];
    _Float16* d = &As[r * KPAD + j * 4];
    d[0] = (_Float16)w.x; d[1] = (_Float16)w.y;
    d[2] = (_Float16)w.z; d[3] = (_Float16)w.w;
  }
  // entity part: 64 rows x 32 float4
  for (int idx = t; idx < TM * 32; idx += 256) {
    int r = idx >> 5, j = idx & 31;
    const float4 w = *(const float4*)&ent[(long)(rowBase + r) * ENT_DIM + j * 4];
    _Float16* d = &As[r * KPAD + WORD_DIM + j * 4];
    d[0] = (_Float16)w.x; d[1] = (_Float16)w.y;
    d[2] = (_Float16)w.z; d[3] = (_Float16)w.w;
  }
  // zero pad K = [428, 448)
  for (int idx = t; idx < TM * 20; idx += 256) {
    int r = idx / 20, j = idx - r * 20;
    As[r * KPAD + (WORD_DIM + ENT_DIM) + j] = (_Float16)0.f;
  }
  __syncthreads();

  const int wave = t >> 5, lane = t & 31;
  const int mRow = lane & 15, kSel = lane >> 4;
  const int n0 = wave * 32;
  v8f c[4][2] = {};
  for (int k0 = 0; k0 < KPAD; k0 += 32) {
    // B-fragment: lane holds N = n + (lane&15), K chunk = k0 + (lane>>4)*16 (32B contiguous)
    v16h b0 = *(const v16h*)&WcT[(size_t)(n0 + mRow) * KPAD + k0 + kSel * 16];
    v16h b1 = *(const v16h*)&WcT[(size_t)(n0 + 16 + mRow) * KPAD + k0 + kSel * 16];
#pragma unroll
    for (int mt = 0; mt < 4; ++mt) {
      v16h a = make_a_frag(&As[(mt * 16 + mRow) * KPAD + k0 + kSel * 16]);
      c[mt][0] = __builtin_amdgcn_wmma_f32_16x16x32_f16(false, a, false, b0, (short)0, c[mt][0], false, false);
      c[mt][1] = __builtin_amdgcn_wmma_f32_16x16x32_f16(false, a, false, b1, (short)0, c[mt][1], false, false);
    }
  }
  // C/D layout: lane element v -> (M = v + 8*(lane>>4), N = lane&15)
  {
    int n  = n0 + mRow;       float bn  = bias[n];
    int n2 = n0 + 16 + mRow;  float bn2 = bias[n2];
#pragma unroll
    for (int mt = 0; mt < 4; ++mt) {
      size_t rb = (size_t)(rowBase + mt * 16 + 8 * kSel);
#pragma unroll
      for (int v = 0; v < 8; ++v) {
        float x = c[mt][0][v] + bn;  x = x > 0.f ? x : 0.f;
        out[(rb + v) * ATT_DIM + n]  = (_Float16)x;
        float y = c[mt][1][v] + bn2; y = y > 0.f ? y : 0.f;
        out[(rb + v) * ATT_DIM + n2] = (_Float16)y;
      }
    }
  }
}

// ---------------------------------------------------------------------------
// Kernel 3: per-row cosine factor = mask(tok) / (||row|| + 1e-9). 1 wave / row.
// ---------------------------------------------------------------------------
__global__ void fac_kernel(const _Float16* __restrict__ mat, const int* __restrict__ tok,
                           float* __restrict__ fac) {
  int t = threadIdx.x;
  long row = (long)blockIdx.x * 8 + (t >> 5);
  int lane = t & 31;
  v8h x = *(const v8h*)&mat[row * ATT_DIM + lane * 8];
  float s = 0.f;
#pragma unroll
  for (int i = 0; i < 8; ++i) { float f = (float)x[i]; s += f * f; }
#pragma unroll
  for (int m = 1; m < 32; m <<= 1) s += __shfl_xor(s, m, 32);
  if (lane == 0)
    fac[row] = (tok[row] == -1) ? 0.f : 1.f / (sqrtf(s) + 1e-9f);
}

// ---------------------------------------------------------------------------
// Kernel 4: cosine-sim WMMA + 11-RBF bank + masked log-sum.
// One workgroup per (batch, q-tile of 16); each wave sweeps 256 docs.
// q-tile A-fragments are loop-invariant => hoisted into registers (8 x v16h).
// q tile staged to LDS via async-to-LDS (ASYNCcnt path) when available.
// ---------------------------------------------------------------------------
__global__ void __launch_bounds__(256) simkern_kernel(
    const _Float16* __restrict__ qmat, const _Float16* __restrict__ dmat,
    const float* __restrict__ qfac, const float* __restrict__ dfac,
    float* __restrict__ partial) {
  const float MU[NKERN]   = {-0.9f,-0.7f,-0.5f,-0.3f,-0.1f,0.1f,0.3f,0.5f,0.7f,0.9f,1.0f};
  const float COEF[NKERN] = {-50.f,-50.f,-50.f,-50.f,-50.f,-50.f,
                             -50.f,-50.f,-50.f,-50.f,-500000.f};  // -0.5/sigma^2
  __shared__ _Float16 Qs[16 * ATT_DIM];          // 8 KB q tile
  __shared__ float QF[16];
  __shared__ float Acc[16][NKERN + 1];           // per-q-row kernel sums + simsum

  const int b = blockIdx.x >> 1, qt = blockIdx.x & 1;
  const int t = threadIdx.x;
  const long qrowBase = (long)b * QLEN + qt * 16;
  {
    int r = t >> 4, seg = t & 15;                // each thread: 32B of the q tile
    const _Float16* g = &qmat[(qrowBase + r) * ATT_DIM + seg * 16];
    _Float16* l = &Qs[r * ATT_DIM + seg * 16];
#if __has_builtin(__builtin_amdgcn_global_load_async_to_lds_b128)
    // builtin takes int4* (generic AS) per hipcc diagnostic: (global src, lds dst, off, cpol)
    __builtin_amdgcn_global_load_async_to_lds_b128((v4i*)g,       (v4i*)l,       0, 0);
    __builtin_amdgcn_global_load_async_to_lds_b128((v4i*)(g + 8), (v4i*)(l + 8), 0, 0);
#if __has_builtin(__builtin_amdgcn_s_wait_asynccnt)
    __builtin_amdgcn_s_wait_asynccnt(0);
#else
    asm volatile("s_wait_asynccnt 0" ::: "memory");
#endif
#else
    *(v16h*)l = *(const v16h*)g;                 // sync fallback
#endif
  }
  if (t < 16) QF[t] = qfac[qrowBase + t];
  if (t < 16 * (NKERN + 1)) ((float*)Acc)[t] = 0.f;
  __syncthreads();

  const int wave = t >> 5, lane = t & 31;
  const int mRow = lane & 15, kSel = lane >> 4;
  const long dbase = (long)b * DLEN;

  float qf[8];
#pragma unroll
  for (int v = 0; v < 8; ++v) qf[v] = QF[v + 8 * kSel];

  // hoist loop-invariant A fragments (q rows) into registers
  v16h aF[8];
#pragma unroll
  for (int kk = 0; kk < 8; ++kk)
    aF[kk] = make_a_frag(&Qs[mRow * ATT_DIM + kk * 32 + kSel * 16]);

  float kacc[NKERN][8];
  float ssum[8];
#pragma unroll
  for (int v = 0; v < 8; ++v) {
    ssum[v] = 0.f;
#pragma unroll
    for (int k = 0; k < NKERN; ++k) kacc[k][v] = 0.f;
  }

  for (int tt = 0; tt < 16; ++tt) {
    const long n0 = (long)wave * 256 + tt * 16;
    const _Float16* drow = &dmat[(dbase + n0 + mRow) * ATT_DIM];
    if (tt < 15) __builtin_prefetch(drow + 16 * ATT_DIM, 0, 1);  // global_prefetch_b8
    v8f c = {};
#pragma unroll
    for (int kk = 0; kk < 8; ++kk) {
      v16h bb = *(const v16h*)&drow[kk * 32 + kSel * 16];
      c = __builtin_amdgcn_wmma_f32_16x16x32_f16(false, aF[kk], false, bb, (short)0, c, false, false);
    }
    float df = dfac[dbase + n0 + mRow];
#pragma unroll
    for (int v = 0; v < 8; ++v) {
      float sim = c[v] * qf[v] * df;     // masked cosine; masked -> 0 (kept in sums, per ref)
      ssum[v] += sim;
#pragma unroll
      for (int k = 0; k < NKERN; ++k) {
        float dd = sim - MU[k];
        kacc[k][v] += __expf(COEF[k] * dd * dd);
      }
    }
  }

  // reduce over the 16 doc-lanes in each half-wave (same q-row group)
#pragma unroll
  for (int v = 0; v < 8; ++v) {
#pragma unroll
    for (int m = 1; m < 16; m <<= 1) ssum[v] += __shfl_xor(ssum[v], m, 32);
#pragma unroll
    for (int k = 0; k < NKERN; ++k) {
#pragma unroll
      for (int m = 1; m < 16; m <<= 1) kacc[k][v] += __shfl_xor(kacc[k][v], m, 32);
    }
  }
  if (mRow == 0) {           // lanes 0 and 16 hold the two q-row halves
#pragma unroll
    for (int v = 0; v < 8; ++v) {
      int qi = v + 8 * kSel;
      atomicAdd(&Acc[qi][NKERN], ssum[v]);       // ds_add_f32
#pragma unroll
      for (int k = 0; k < NKERN; ++k) atomicAdd(&Acc[qi][k], kacc[k][v]);
    }
  }
  __syncthreads();

  if (t < NKERN) {
    float s = 0.f;
    for (int qi = 0; qi < 16; ++qi) {
      if (Acc[qi][NKERN] != 0.f) s += logf(Acc[qi][t] + 1e-6f);
    }
    partial[blockIdx.x * NKERN + t] = s;
  }
}

// ---------------------------------------------------------------------------
// Kernel 5: scores = result @ W_c + b_c
// ---------------------------------------------------------------------------
__global__ void final_kernel(const float* __restrict__ partial, const float* __restrict__ Wc,
                             const float* __restrict__ bc, float* __restrict__ out) {
  int b = threadIdx.x;
  if (b >= B_SZ) return;
  float s = bc[0];
#pragma unroll
  for (int k = 0; k < NKERN; ++k)
    s += (partial[(2 * b) * NKERN + k] + partial[(2 * b + 1) * NKERN + k]) * Wc[k];
  out[b] = s;
}

// ---------------------------------------------------------------------------
// Workspace layout (bytes, all offsets 256B-aligned):
//   [0)            WcT   f16  256*448*2 = 229376
//   [229376)       bias  f32  1024
//   [230400)       qmat  f16  64*32*256*2   = 1048576
//   [1278976)      dmat  f16  64*2048*256*2 = 67108864
//   [68387840)     qfac  f32  8192
//   [68396032)     dfac  f32  524288
//   [68920320)     part  f32  128*11*4 = 5632       (total ~65.7 MiB)
// ---------------------------------------------------------------------------
extern "C" void kernel_launch(void* const* d_in, const int* in_sizes, int n_in,
                              void* d_out, int out_size, void* d_ws, size_t ws_size,
                              hipStream_t stream) {
  (void)in_sizes; (void)n_in; (void)out_size; (void)ws_size;
  const int*   query_tok = (const int*)d_in[0];
  const int*   doc_tok   = (const int*)d_in[1];
  const float* query_ent = (const float*)d_in[2];
  const float* doc_ent   = (const float*)d_in[3];
  const float* embed     = (const float*)d_in[4];
  const float* Wt        = (const float*)d_in[5];
  const float* bt        = (const float*)d_in[6];
  const float* We        = (const float*)d_in[7];
  const float* be        = (const float*)d_in[8];
  const float* Wc        = (const float*)d_in[9];
  const float* bc        = (const float*)d_in[10];

  char* ws = (char*)d_ws;
  _Float16* WcT  = (_Float16*)(ws + 0);
  float*    bias = (float*)   (ws + 229376);
  _Float16* qmat = (_Float16*)(ws + 230400);
  _Float16* dmat = (_Float16*)(ws + 1278976);
  float*    qfac = (float*)   (ws + 68387840);
  float*    dfac = (float*)   (ws + 68396032);
  float*    part = (float*)   (ws + 68920320);
  float*    out  = (float*)d_out;

  prep_weights<<<KPAD, 256, 0, stream>>>(Wt, bt, We, be, WcT, bias);
  transform_kernel<<<(B_SZ * DLEN) / TM, 256, 0, stream>>>(doc_tok, doc_ent, embed, WcT, bias, dmat);
  transform_kernel<<<(B_SZ * QLEN) / TM, 256, 0, stream>>>(query_tok, query_ent, embed, WcT, bias, qmat);
  fac_kernel<<<(B_SZ * DLEN) / 8, 256, 0, stream>>>(dmat, doc_tok, dfac);
  fac_kernel<<<(B_SZ * QLEN) / 8, 256, 0, stream>>>(qmat, query_tok, qfac);
  simkern_kernel<<<B_SZ * 2, 256, 0, stream>>>(qmat, dmat, qfac, dfac, part);
  final_kernel<<<1, 64, 0, stream>>>(part, Wc, bc, out);
}
